// CausalSelfAttention_2619930051559
// MI455X (gfx1250) — matrix operand order repaired and checked by reading, not compiled, not run on hardware
//
#include <hip/hip_runtime.h>
#include <hip/hip_bf16.h>
#include <math.h>

typedef __attribute__((ext_vector_type(16))) _Float16 v16h;
typedef __attribute__((ext_vector_type(8)))  _Float16 v8h;
typedef __attribute__((ext_vector_type(2)))  _Float16 v2h;
typedef __attribute__((ext_vector_type(8)))  float    v8f;

__device__ __forceinline__ v8f wmma_f16(v16h a, v16h b, v8f c) {
    return __builtin_amdgcn_wmma_f32_16x16x32_f16(false, a, false, b, (short)0, c, false, false);
}

__device__ __forceinline__ v16h cat16(v8h lo, v8h hi) {
    v16h r;
#pragma unroll
    for (int i = 0; i < 8; ++i) { r[i] = lo[i]; r[i + 8] = hi[i]; }
    return r;
}

// load 8 contiguous f32 (16B-aligned) -> 8 f16
__device__ __forceinline__ v8h cvt8(const float* __restrict__ p) {
    float4 f0 = *(const float4*)p;
    float4 f1 = *(const float4*)(p + 4);
    v8h r;
    r[0] = (_Float16)f0.x; r[1] = (_Float16)f0.y; r[2] = (_Float16)f0.z; r[3] = (_Float16)f0.w;
    r[4] = (_Float16)f1.x; r[5] = (_Float16)f1.y; r[6] = (_Float16)f1.z; r[7] = (_Float16)f1.w;
    return r;
}

// ---------------------------------------------------------------------------
// GEMM: C[M,N] = A[M,K] * B[K,N], fp32 in/out, f16 WMMA compute.
// Block tile 128x128, K-step 32. 256 threads = 8 waves in a 4(M) x 2(N) grid,
// each wave computes a 32x64 patch = 2x4 WMMA tiles (64 WMMA / WG / K-step).
// sB is staged transposed [n][k]; each thread owns 2 consecutive k at 8 n so
// transposed writes are packed 2xf16 (ds_store_b32).
// ---------------------------------------------------------------------------
__global__ __launch_bounds__(256)
void gemm_wmma_f16(const float* __restrict__ A, const float* __restrict__ B,
                   float* __restrict__ C, int M, int N, int K) {
    __shared__ _Float16 sA[128 * 32];  // [m][k], k contiguous
    __shared__ _Float16 sB[128 * 32];  // [n][k] (transposed), k contiguous

    const int tid    = threadIdx.x;
    const int wid    = tid >> 5;
    const int lane   = tid & 31;
    const int g      = lane >> 4;      // half-wave group
    const int hm     = lane & 15;
    const int wave_m = wid & 3;        // 0..3 -> 32-row chunks
    const int wave_n = wid >> 2;       // 0..1 -> 64-col chunks

    const int row0 = blockIdx.y * 128;
    const int col0 = blockIdx.x * 128;

    v8f acc[2][4] = {};

    const int a_row = tid >> 1;            // 0..127
    const int a_k   = (tid & 1) * 16;      // 0 or 16
    const int b_kg  = tid >> 4;            // 0..15 -> k = 2*b_kg, 2*b_kg+1
    const int b_n8  = (tid & 15) * 8;      // 0..120

    for (int kk = 0; kk < K; kk += 32) {
        // --- cooperative staging (f32 global -> f16 LDS) ---
        {
            const float* ap = A + (size_t)(row0 + a_row) * K + kk + a_k;
#pragma unroll
            for (int c = 0; c < 4; ++c) {
                float4 f = *(const float4*)(ap + c * 4);
                _Float16* d = &sA[a_row * 32 + a_k + c * 4];
                d[0] = (_Float16)f.x; d[1] = (_Float16)f.y;
                d[2] = (_Float16)f.z; d[3] = (_Float16)f.w;
            }
            const float* bp0 = B + (size_t)(kk + 2 * b_kg) * N + col0 + b_n8;
            const float* bp1 = bp0 + N;
#pragma unroll
            for (int c = 0; c < 2; ++c) {
                float4 f0 = *(const float4*)(bp0 + c * 4);
                float4 f1 = *(const float4*)(bp1 + c * 4);
                const float* a0 = (const float*)&f0;
                const float* a1 = (const float*)&f1;
#pragma unroll
                for (int i = 0; i < 4; ++i) {
                    v2h p;
                    p[0] = (_Float16)a0[i];
                    p[1] = (_Float16)a1[i];
                    *(v2h*)&sB[(b_n8 + c * 4 + i) * 32 + 2 * b_kg] = p;
                }
            }
        }
        if (kk + 32 < K) {
            __builtin_prefetch(A + (size_t)(row0 + a_row) * K + kk + 32 + a_k, 0, 0);
            __builtin_prefetch(B + (size_t)(kk + 32 + 2 * b_kg) * N + col0 + b_n8, 0, 0);
        }
        __syncthreads();

        // --- fragments from LDS ---
        v16h af[2], bf[4];
#pragma unroll
        for (int tm = 0; tm < 2; ++tm) {
            const int m = wave_m * 32 + tm * 16 + hm;
            af[tm] = cat16(*(const v8h*)&sA[m * 32 + 8 * g],
                           *(const v8h*)&sA[m * 32 + 16 + 8 * g]);
        }
#pragma unroll
        for (int tn = 0; tn < 4; ++tn) {
            const int n = wave_n * 64 + tn * 16 + hm;
            bf[tn] = cat16(*(const v8h*)&sB[n * 32 + 8 * g],
                           *(const v8h*)&sB[n * 32 + 16 + 8 * g]);
        }
#pragma unroll
        for (int tm = 0; tm < 2; ++tm)
#pragma unroll
            for (int tn = 0; tn < 4; ++tn)
                acc[tm][tn] = wmma_f16(af[tm], bf[tn], acc[tm][tn]);
        __syncthreads();
    }

    // --- store: acc element r -> row r+8g, col hm ---
#pragma unroll
    for (int tm = 0; tm < 2; ++tm)
#pragma unroll
        for (int tn = 0; tn < 4; ++tn) {
            const int col = col0 + wave_n * 64 + tn * 16 + hm;
#pragma unroll
            for (int r = 0; r < 8; ++r) {
                const int row = row0 + wave_m * 32 + tm * 16 + r + 8 * g;
                C[(size_t)row * N + col] = acc[tm][tn][r];
            }
        }
}

// ---------------------------------------------------------------------------
// Flash attention: qkv (B,T,3C) row-major; per (b, h, 128-query tile).
// 8 waves x 16 query rows. Key blocks of 32. Online softmax in fp32.
// scale(t) = (1/8) * log(t+1)^2, causal mask.
// ---------------------------------------------------------------------------
#define T_SEQ 2048
#define C_DIM 1024
#define H_NUM 16
#define D_HEAD 64
#define C3 3072

__global__ __launch_bounds__(256)
void flash_attn_wmma(const float* __restrict__ qkv, float* __restrict__ y) {
    __shared__ _Float16 sK[32 * 64];        // [key][d]
    __shared__ _Float16 sVt[64 * 32];       // [d][key] (transposed)
    __shared__ _Float16 sP[8 * 16 * 32];    // per-wave 16x32 P scratch

    const int tid  = threadIdx.x;
    const int wid  = tid >> 5;
    const int lane = tid & 31;
    const int g    = lane >> 4;
    const int hm   = lane & 15;

    const int q0 = blockIdx.x * 128;
    const int h  = blockIdx.y;
    const int b  = blockIdx.z;

    // ---- load Q fragments (16 rows x 64 dims = 2 A-fragments) ----
    v16h aq[2];
    {
        const int t = q0 + wid * 16 + hm;
        const float* qp = qkv + ((size_t)(b * T_SEQ + t)) * C3 + h * D_HEAD;
#pragma unroll
        for (int kc = 0; kc < 2; ++kc)
            aq[kc] = cat16(cvt8(qp + kc * 32 + 8 * g),
                           cvt8(qp + kc * 32 + 16 + 8 * g));
    }

    // per-lane per-row state (row = r + 8*g of this wave's 16-row tile)
    float mrow[8], lrow[8], rowscale[8];
    int   trow[8];
#pragma unroll
    for (int r = 0; r < 8; ++r) {
        trow[r] = q0 + wid * 16 + r + 8 * g;
        const float lg = __logf((float)(trow[r] + 1));
        rowscale[r] = 0.125f * lg * lg;
        mrow[r] = -INFINITY;
        lrow[r] = 0.0f;
    }
    v8f oacc[4] = {};

    const int k_key = tid >> 3;            // 0..31   (sK staging)
    const int k_d8  = (tid & 7) * 8;       // 0..56
    const int v_kp  = (tid >> 4) * 2;      // 0..30   (sVt staging, key pairs)
    const int v_d4  = (tid & 15) * 4;      // 0..60
    const int nblocks = 4 * (blockIdx.x + 1);

    for (int j = 0; j < nblocks; ++j) {
        const int kbase = j * 32;
        // ---- stage K tile [key][d] (contiguous f16 writes -> b128) ----
        {
            const float* kp = qkv + ((size_t)(b * T_SEQ + kbase + k_key)) * C3
                              + C_DIM + h * D_HEAD + k_d8;
#pragma unroll
            for (int c = 0; c < 2; ++c) {
                float4 f = *(const float4*)(kp + c * 4);
                _Float16* d = &sK[k_key * 64 + k_d8 + c * 4];
                d[0] = (_Float16)f.x; d[1] = (_Float16)f.y;
                d[2] = (_Float16)f.z; d[3] = (_Float16)f.w;
            }
        }
        // ---- stage V^T tile [d][key] (packed key-pairs -> b32) ----
        {
            const float* vp0 = qkv + ((size_t)(b * T_SEQ + kbase + v_kp)) * C3
                               + 2 * C_DIM + h * D_HEAD + v_d4;
            const float* vp1 = vp0 + C3;
            float4 f0 = *(const float4*)vp0;
            float4 f1 = *(const float4*)vp1;
            const float* a0 = (const float*)&f0;
            const float* a1 = (const float*)&f1;
#pragma unroll
            for (int i = 0; i < 4; ++i) {
                v2h p;
                p[0] = (_Float16)a0[i];
                p[1] = (_Float16)a1[i];
                *(v2h*)&sVt[(v_d4 + i) * 32 + v_kp] = p;
            }
        }
        __syncthreads();

        // ---- S = Q * K^T  (16 x 32 scores, two 16x16 halves) ----
        v8f s[2] = {};
#pragma unroll
        for (int half = 0; half < 2; ++half) {
            const int n = half * 16 + hm;  // key index in tile
#pragma unroll
            for (int kc = 0; kc < 2; ++kc) {
                v16h bk = cat16(*(const v8h*)&sK[n * 64 + kc * 32 + 8 * g],
                                *(const v8h*)&sK[n * 64 + kc * 32 + 16 + 8 * g]);
                s[half] = wmma_f16(aq[kc], bk, s[half]);
            }
        }

        // ---- scale + causal mask + online softmax ----
        float p[2][8];
#pragma unroll
        for (int r = 0; r < 8; ++r) {
            float v0 = s[0][r] * rowscale[r];
            float v1 = s[1][r] * rowscale[r];
            if (kbase + hm      > trow[r]) v0 = -INFINITY;
            if (kbase + 16 + hm > trow[r]) v1 = -INFINITY;
            float rmax = fmaxf(v0, v1);
#pragma unroll
            for (int off = 1; off < 16; off <<= 1)
                rmax = fmaxf(rmax, __shfl_xor(rmax, off, 32));
            const float mnew  = fmaxf(mrow[r], rmax);
            const float scale = __expf(mrow[r] - mnew);
            const float p0 = __expf(v0 - mnew);
            const float p1 = __expf(v1 - mnew);
            float rsum = p0 + p1;
#pragma unroll
            for (int off = 1; off < 16; off <<= 1)
                rsum += __shfl_xor(rsum, off, 32);
            lrow[r] = lrow[r] * scale + rsum;
            mrow[r] = mnew;
            p[0][r] = p0;
            p[1][r] = p1;
#pragma unroll
            for (int c = 0; c < 4; ++c) oacc[c][r] *= scale;
        }

        // ---- P through wave-private LDS to build A-fragment ----
        _Float16* myP = &sP[wid * 512];
#pragma unroll
        for (int half = 0; half < 2; ++half)
#pragma unroll
            for (int r = 0; r < 8; ++r)
                myP[(r + 8 * g) * 32 + half * 16 + hm] = (_Float16)p[half][r];
        const v16h pa = cat16(*(const v8h*)&myP[hm * 32 + 8 * g],
                              *(const v8h*)&myP[hm * 32 + 16 + 8 * g]);

        // ---- O += P * V ----
#pragma unroll
        for (int c = 0; c < 4; ++c) {
            v16h bv = cat16(*(const v8h*)&sVt[(c * 16 + hm) * 32 + 8 * g],
                            *(const v8h*)&sVt[(c * 16 + hm) * 32 + 16 + 8 * g]);
            oacc[c] = wmma_f16(pa, bv, oacc[c]);
        }
        __syncthreads();
    }

    // ---- normalize and store y (B,T,C) ----
    float inv[8];
#pragma unroll
    for (int r = 0; r < 8; ++r) inv[r] = 1.0f / lrow[r];
#pragma unroll
    for (int c = 0; c < 4; ++c)
#pragma unroll
        for (int r = 0; r < 8; ++r)
            y[((size_t)(b * T_SEQ) + trow[r]) * C_DIM + h * D_HEAD + c * 16 + hm]
                = oacc[c][r] * inv[r];
}

// ---------------------------------------------------------------------------
extern "C" void kernel_launch(void* const* d_in, const int* in_sizes, int n_in,
                              void* d_out, int out_size, void* d_ws, size_t ws_size,
                              hipStream_t stream) {
    const float* x      = (const float*)d_in[0];  // (4,2048,1024)
    const float* w_attn = (const float*)d_in[1];  // (1024,3072)
    const float* w_proj = (const float*)d_in[2];  // (1024,1024)
    float* out = (float*)d_out;                   // (4,2048,1024)

    float* qkv = (float*)d_ws;                    // 8192 x 3072
    float* yws = qkv + (size_t)8192 * 3072;       // 8192 x 1024

    dim3 blk(256);

    // 1) qkv = x @ w_attn : M=8192 N=3072 K=1024
    gemm_wmma_f16<<<dim3(3072 / 128, 8192 / 128), blk, 0, stream>>>(
        x, w_attn, qkv, 8192, 3072, 1024);

    // 2) flash attention -> y
    flash_attn_wmma<<<dim3(T_SEQ / 128, H_NUM, 4), blk, 0, stream>>>(qkv, yws);

    // 3) out = y @ w_proj : M=8192 N=1024 K=1024
    gemm_wmma_f16<<<dim3(1024 / 128, 8192 / 128), blk, 0, stream>>>(
        yws, w_proj, out, 8192, 1024, 1024);
}
